// EquiFormerBlock_22033182228665
// MI455X (gfx1250) — matrix operand order repaired and checked by reading, not compiled, not run on hardware
//
#include <hip/hip_runtime.h>
#include <math.h>

// ---------------------------------------------------------------------------
// EquiFormer block for MI455X (gfx1250): bf16 WMMA GEMMs + LDS neighbor cache
// + async global->LDS gather; bf16 pack via native casts (backend picks op).
// ---------------------------------------------------------------------------

typedef __attribute__((ext_vector_type(16))) __bf16 v16bf;
typedef __attribute__((ext_vector_type(8)))  float  v8f;
typedef __attribute__((ext_vector_type(4)))  int    v4i;

#define B_    4
#define N_    2048
#define K_    48
#define CSI   128
#define CSO   128
#define NODES (B_*N_)

#if defined(__has_builtin)
#if __has_builtin(__builtin_amdgcn_global_load_async_to_lds_b128) && \
    __has_builtin(__builtin_amdgcn_s_wait_asynccnt)
#define HAVE_ASYNC 1
#endif
#endif

#ifdef HAVE_ASYNC
typedef __attribute__((address_space(1))) v4i* gptr_v4i;   // global int4*
typedef __attribute__((address_space(3))) v4i* lptr_v4i;   // LDS int4*
#endif

// f32 pair -> packed bf16x2 (RNE); let the backend choose the best lowering
__device__ __forceinline__ unsigned int pk2(float a, float b) {
    union { __bf16 h[2]; unsigned int u; } c;
    c.h[0] = (__bf16)a;
    c.h[1] = (__bf16)b;
    return c.u;
}
__device__ __forceinline__ unsigned short f2bf(float f) {
    union { __bf16 h; unsigned short u; } c;
    c.h = (__bf16)f;
    return c.u;
}
__device__ __forceinline__ float gelu_exact(float x) {
    return 0.5f * x * (1.0f + erff(x * 0.70710678118654752440f));
}

union FragBF {
    v16bf v;
    uint4 q[2];
    unsigned short u[16];
};

// ---------------------------------------------------------------------------
// Kernel 0: pack weight matrices (Kdim x Ndim, row-major f32) into bf16
// B-fragment layout: fragment (nt,kc) = 32 lanes x 16 bf16; lane l covers
// col = nt*16 + (l&15), K = kc*32 + (l>>4)*8 + j + (j>=8 ? 8 : 0)  (j = 0..15).
// Fragment blocks: [0,44) W_lin(128x176) | [44,76) W_a1 rows128..255 (128x128)
//                  | [76,204) W_f1(128x512) | [204,332) W_f2(512x128)
// ---------------------------------------------------------------------------
__global__ void pack_weights(const float* __restrict__ Wlin,
                             const float* __restrict__ Wa1,
                             const float* __restrict__ Wf1,
                             const float* __restrict__ Wf2,
                             unsigned short* __restrict__ pkWlin,
                             unsigned short* __restrict__ pkWa1b,
                             unsigned short* __restrict__ pkWf1,
                             unsigned short* __restrict__ pkWf2) {
    int bid  = blockIdx.x;
    int lane = threadIdx.x;          // 32 threads
    const float* src; unsigned short* dst; int KC, ld, koff, fid;
    if (bid < 44)       { fid = bid;       src = Wlin; dst = pkWlin; KC = 4;  ld = 176; koff = 0;   }
    else if (bid < 76)  { fid = bid - 44;  src = Wa1;  dst = pkWa1b; KC = 4;  ld = 128; koff = 128; }
    else if (bid < 204) { fid = bid - 76;  src = Wf1;  dst = pkWf1;  KC = 4;  ld = 512; koff = 0;   }
    else                { fid = bid - 204; src = Wf2;  dst = pkWf2;  KC = 16; ld = 128; koff = 0;   }
    int nt = fid / KC, kc = fid % KC;
    int col = nt * 16 + (lane & 15);
    int kb  = kc * 32 + (lane >> 4) * 8;
    unsigned int* o = (unsigned int*)(dst + ((size_t)fid * 32 + lane) * 16);
    #pragma unroll
    for (int j = 0; j < 16; j += 2) {
        float a  = src[(size_t)(kb + j + (j >= 8 ? 8 : 0) + koff) * ld + col];
        float bb = src[(size_t)(kb + j + 1 + (j >= 8 ? 8 : 0) + koff) * ld + col];
        o[j >> 1] = pk2(a, bb);
    }
}

// ---------------------------------------------------------------------------
// Kernel 1: lin_out = scalar_feats @ W_lin + b_lin  (M=8192,K=128,N=176)
// Split into scalar_out (cols 0..127) and vec_out (cols 128..175).
// One wave per 16x16 tile; 8 waves / block.  5632 wave-tiles -> 704 blocks.
// ---------------------------------------------------------------------------
__global__ __launch_bounds__(256) void k_lin(const float* __restrict__ X,
                                             const unsigned short* __restrict__ pkW,
                                             const float* __restrict__ bias,
                                             float* __restrict__ sOut,
                                             float* __restrict__ vOut) {
    int wave = threadIdx.x >> 5, lane = threadIdx.x & 31;
    int wt = blockIdx.x * 8 + wave;                 // 0..5631
    int mt = wt / 11, nt = wt % 11;
    int h = lane >> 4, lm = lane & 15;
    const float* arow = X + (size_t)(mt * 16 + lm) * CSI;
    v8f acc = {};
    #pragma unroll
    for (int c = 0; c < 4; ++c) {
        FragBF A, Bf;
        const float4* p0 = (const float4*)(arow + c * 32 + h * 8);
        const float4* p1 = (const float4*)(arow + c * 32 + h * 8 + 16);
        float4 x0 = p0[0], x1 = p0[1], y0 = p1[0], y1 = p1[1];
        A.q[0] = make_uint4(pk2(x0.x, x0.y), pk2(x0.z, x0.w), pk2(x1.x, x1.y), pk2(x1.z, x1.w));
        A.q[1] = make_uint4(pk2(y0.x, y0.y), pk2(y0.z, y0.w), pk2(y1.x, y1.y), pk2(y1.z, y1.w));
        const uint4* bp = (const uint4*)(pkW + ((size_t)(nt * 4 + c) * 32 + lane) * 16);
        Bf.q[0] = bp[0]; Bf.q[1] = bp[1];
        acc = __builtin_amdgcn_wmma_f32_16x16x32_bf16(false, A.v, false, Bf.v, (short)0, acc, false, false);
    }
    int col = nt * 16 + lm;
    float bv = bias[col];
    #pragma unroll
    for (int r = 0; r < 8; ++r) {
        int m = mt * 16 + r + 8 * h;
        float vv = acc[r] + bv;
        if (col < CSO) sOut[(size_t)m * CSO + col] = vv;
        else           vOut[(size_t)m * 48 + (col - CSO)] = vv;
    }
}

// ---------------------------------------------------------------------------
// Kernel 2: per-node edge attention.  One 256-thread block per node.
//   z[k,j] = base[j] + (neigh_k @ W_a1[128:256])[j] + d_k * W_a1[256,j]
//   logit[k] = sum_j gelu(z[k,j]) * W_a2[j] + b_a2 ; mask/clip/softmax over K
//   scalar_msg / vec_msg weighted sums (f32, deterministic order).
// ---------------------------------------------------------------------------
__global__ __launch_bounds__(256) void k_att(const float* __restrict__ coords,
                                             const int*   __restrict__ Eidx,
                                             const float* __restrict__ mask,
                                             const float* __restrict__ Wa1,
                                             const float* __restrict__ Wa2,
                                             const float* __restrict__ ba1,
                                             const float* __restrict__ ba2,
                                             const unsigned short* __restrict__ pkWa1b,
                                             const float* __restrict__ sOut,
                                             const float* __restrict__ vOut,
                                             float* __restrict__ xPre,
                                             float* __restrict__ outVec) {
    __shared__ __attribute__((aligned(16))) float          nF[K_ * 128];   // f32 neighbor rows
    __shared__ __attribute__((aligned(16))) unsigned short nB[K_ * 136];   // bf16, padded stride
    __shared__ float central[128], base_s[128], w1c_s[128], w2_s[128];
    __shared__ float d_s[K_], att_s[K_], logit_s[K_], msk_s[K_];
    __shared__ int   idx_s[K_];
    __shared__ float part[8][K_];

    int node = blockIdx.x;
    int b = node / N_;
    int t = threadIdx.x, lane = t & 31, wave = t >> 5;

    if (t < K_) {
        int e = Eidx[(size_t)node * K_ + t];
        idx_s[t] = e;
        int nb = b * N_ + e;
        msk_s[t] = mask[nb];
        float dx = coords[(size_t)node * 3 + 0] - coords[(size_t)nb * 3 + 0];
        float dy = coords[(size_t)node * 3 + 1] - coords[(size_t)nb * 3 + 1];
        float dz = coords[(size_t)node * 3 + 2] - coords[(size_t)nb * 3 + 2];
        d_s[t] = sqrtf(dx * dx + dy * dy + dz * dz + 1e-6f);
    }
    if (t < 128) central[t] = sOut[(size_t)node * 128 + t];
    __syncthreads();

    // gather 6 neighbor rows per wave into LDS (f32), then bf16 copies
#ifdef HAVE_ASYNC
    #pragma unroll
    for (int kk0 = 0; kk0 < 6; ++kk0) {
        int kk = wave * 6 + kk0;
        int nb = b * N_ + idx_s[kk];
        gptr_v4i src = (gptr_v4i)(uintptr_t)(sOut + (size_t)nb * 128 + lane * 4);
        lptr_v4i dst = (lptr_v4i)(unsigned int)(uintptr_t)(nF + kk * 128 + lane * 4);
        __builtin_amdgcn_global_load_async_to_lds_b128(src, dst, 0, 0);
    }
    __builtin_amdgcn_s_wait_asynccnt(0);
    #pragma unroll
    for (int kk0 = 0; kk0 < 6; ++kk0) {
        int kk = wave * 6 + kk0;
        float4 v = ((const float4*)(nF + kk * 128))[lane];
        unsigned int* dst = (unsigned int*)(nB + kk * 136);
        dst[lane * 2]     = pk2(v.x, v.y);
        dst[lane * 2 + 1] = pk2(v.z, v.w);
    }
#else
    for (int kk = wave * 6; kk < wave * 6 + 6; ++kk) {
        int nb = b * N_ + idx_s[kk];
        float4 v = ((const float4*)(sOut + (size_t)nb * 128))[lane];
        ((float4*)(nF + kk * 128))[lane] = v;
        unsigned int* dst = (unsigned int*)(nB + kk * 136);
        dst[lane * 2]     = pk2(v.x, v.y);
        dst[lane * 2 + 1] = pk2(v.z, v.w);
    }
#endif
    // base[j] = central @ W_a1[0:128] + b_a1 ; cache d-column & W_a2
    if (t < 128) {
        float acc = ba1[t];
        for (int c = 0; c < 128; ++c) acc += central[c] * Wa1[c * 128 + t];
        base_s[t] = acc;
        w1c_s[t]  = Wa1[256 * 128 + t];
        w2_s[t]   = Wa2[t];
    }
    __syncthreads();

    // WMMA: 48x128x128 per node. wave -> N-tile, 3 M-tiles, 4 K-chunks.
    int h = lane >> 4, lm = lane & 15;
    v8f accs[3] = {};
    #pragma unroll
    for (int c = 0; c < 4; ++c) {
        FragBF Bf;
        const uint4* bp = (const uint4*)(pkWa1b + ((size_t)(wave * 4 + c) * 32 + lane) * 16);
        Bf.q[0] = bp[0]; Bf.q[1] = bp[1];
        #pragma unroll
        for (int mt = 0; mt < 3; ++mt) {
            FragBF A;
            const unsigned short* ar = nB + (mt * 16 + lm) * 136 + c * 32 + h * 8;
            A.q[0] = *(const uint4*)ar;
            A.q[1] = *(const uint4*)(ar + 16);
            accs[mt] = __builtin_amdgcn_wmma_f32_16x16x32_bf16(false, A.v, false, Bf.v,
                                                               (short)0, accs[mt], false, false);
        }
    }

    // gelu + @W_a2 ; 16-lane tree reduce over j, deterministic wave partials
    int j = wave * 16 + lm;
    float basej = base_s[j], w1cj = w1c_s[j], w2j = w2_s[j];
    #pragma unroll
    for (int mt = 0; mt < 3; ++mt) {
        #pragma unroll
        for (int r = 0; r < 8; ++r) {
            int k = mt * 16 + r + 8 * h;
            float z = accs[mt][r] + basej + d_s[k] * w1cj;
            float contrib = gelu_exact(z) * w2j;
            contrib += __shfl_xor(contrib, 1, 16);
            contrib += __shfl_xor(contrib, 2, 16);
            contrib += __shfl_xor(contrib, 4, 16);
            contrib += __shfl_xor(contrib, 8, 16);
            if (lm == 0) part[wave][k] = contrib;   // each (wave,k) written once
        }
    }
    __syncthreads();

    if (t < K_) {
        float v = ba2[0];
        for (int w = 0; w < 8; ++w) v += part[w][t];
        if (msk_s[t] == 0.0f) v = -10000.0f;
        v = fminf(fmaxf(v, -10000.0f), 10.0f);
        logit_s[t] = v;
    }
    __syncthreads();
    if (t == 0) {
        float mx = -1e30f;
        for (int k = 0; k < K_; ++k) mx = fmaxf(mx, logit_s[k]);
        float s = 0.0f;
        for (int k = 0; k < K_; ++k) { float e = expf(logit_s[k] - mx); att_s[k] = e; s += e; }
        float inv = 1.0f / s;
        for (int k = 0; k < K_; ++k) att_s[k] *= inv;
    }
    __syncthreads();

    if (t < 128) {
        float sm = 0.0f;
        for (int k = 0; k < K_; ++k) sm += att_s[k] * nF[k * 128 + t];
        xPre[(size_t)node * 128 + t] = central[t] + sm;           // residual pre-LN
    } else if (t < 176) {
        int jj = t - 128;
        float vm = 0.0f;
        for (int k = 0; k < K_; ++k)
            vm += att_s[k] * vOut[(size_t)(b * N_ + idx_s[k]) * 48 + jj];
        outVec[(size_t)node * 48 + jj] = vm;
    }
}

// ---------------------------------------------------------------------------
// Kernel 3: LayerNorm -> xn (f32 for residual, bf16 for FFN GEMM)
// ---------------------------------------------------------------------------
__global__ __launch_bounds__(128) void k_ln(const float* __restrict__ xPre,
                                            const float* __restrict__ g,
                                            const float* __restrict__ bta,
                                            float* __restrict__ xnF,
                                            unsigned short* __restrict__ xnB) {
    __shared__ float red[128];
    int node = blockIdx.x, t = threadIdx.x;
    float x = xPre[(size_t)node * 128 + t];
    red[t] = x; __syncthreads();
    for (int s = 64; s > 0; s >>= 1) { if (t < s) red[t] += red[t + s]; __syncthreads(); }
    float mu = red[0] * (1.0f / 128.0f);
    __syncthreads();
    float dv = x - mu;
    red[t] = dv * dv; __syncthreads();
    for (int s = 64; s > 0; s >>= 1) { if (t < s) red[t] += red[t + s]; __syncthreads(); }
    float var = red[0] * (1.0f / 128.0f);
    float xn = dv * rsqrtf(var + 1e-5f) * g[t] + bta[t];
    xnF[(size_t)node * 128 + t] = xn;
    xnB[(size_t)node * 128 + t] = f2bf(xn);
}

// ---------------------------------------------------------------------------
// Kernel 4: g = gelu(xn @ W_f1 + b_f1)   (M=8192,K=128,N=512) -> bf16
// ---------------------------------------------------------------------------
__global__ __launch_bounds__(256) void k_ffn1(const unsigned short* __restrict__ xnB,
                                              const unsigned short* __restrict__ pkWf1,
                                              const float* __restrict__ bf1,
                                              unsigned short* __restrict__ gOut) {
    int wave = threadIdx.x >> 5, lane = threadIdx.x & 31;
    int wt = blockIdx.x * 8 + wave;            // 0..16383
    int mt = wt >> 5, nt = wt & 31;
    int h = lane >> 4, lm = lane & 15;
    const unsigned short* arow = xnB + (size_t)(mt * 16 + lm) * 128;
    v8f acc = {};
    #pragma unroll
    for (int c = 0; c < 4; ++c) {
        FragBF A, Bf;
        const unsigned short* ar = arow + c * 32 + h * 8;
        A.q[0] = *(const uint4*)ar;
        A.q[1] = *(const uint4*)(ar + 16);
        const uint4* bp = (const uint4*)(pkWf1 + ((size_t)(nt * 4 + c) * 32 + lane) * 16);
        Bf.q[0] = bp[0]; Bf.q[1] = bp[1];
        acc = __builtin_amdgcn_wmma_f32_16x16x32_bf16(false, A.v, false, Bf.v, (short)0, acc, false, false);
    }
    int col = nt * 16 + lm;
    float bv = bf1[col];
    #pragma unroll
    for (int r = 0; r < 8; ++r) {
        int m = mt * 16 + r + 8 * h;
        gOut[(size_t)m * 512 + col] = f2bf(gelu_exact(acc[r] + bv));
    }
}

// ---------------------------------------------------------------------------
// Kernel 5: out_x = xn + (g @ W_f2 + b_f2)  (M=8192,K=512,N=128)
// ---------------------------------------------------------------------------
__global__ __launch_bounds__(256) void k_ffn2(const unsigned short* __restrict__ gIn,
                                              const unsigned short* __restrict__ pkWf2,
                                              const float* __restrict__ bf2,
                                              const float* __restrict__ xnF,
                                              float* __restrict__ outX) {
    int wave = threadIdx.x >> 5, lane = threadIdx.x & 31;
    int wt = blockIdx.x * 8 + wave;            // 0..4095
    int mt = wt >> 3, nt = wt & 7;
    int h = lane >> 4, lm = lane & 15;
    const unsigned short* arow = gIn + (size_t)(mt * 16 + lm) * 512;
    v8f acc = {};
    #pragma unroll
    for (int c = 0; c < 16; ++c) {
        FragBF A, Bf;
        const unsigned short* ar = arow + c * 32 + h * 8;
        A.q[0] = *(const uint4*)ar;
        A.q[1] = *(const uint4*)(ar + 16);
        const uint4* bp = (const uint4*)(pkWf2 + ((size_t)(nt * 16 + c) * 32 + lane) * 16);
        Bf.q[0] = bp[0]; Bf.q[1] = bp[1];
        acc = __builtin_amdgcn_wmma_f32_16x16x32_bf16(false, A.v, false, Bf.v, (short)0, acc, false, false);
    }
    int col = nt * 16 + lm;
    float bv = bf2[col];
    #pragma unroll
    for (int r = 0; r < 8; ++r) {
        int m = mt * 16 + r + 8 * h;
        outX[(size_t)m * 128 + col] = xnF[(size_t)m * 128 + col] + acc[r] + bv;
    }
}

// ---------------------------------------------------------------------------
// Launcher
// ---------------------------------------------------------------------------
extern "C" void kernel_launch(void* const* d_in, const int* in_sizes, int n_in,
                              void* d_out, int out_size, void* d_ws, size_t ws_size,
                              hipStream_t stream) {
    const float* scalar_feats = (const float*)d_in[0];
    const float* coords       = (const float*)d_in[1];
    const int*   E_idx        = (const int*)  d_in[2];
    const float* mask         = (const float*)d_in[3];
    const float* W_lin        = (const float*)d_in[4];
    const float* b_lin        = (const float*)d_in[5];
    const float* W_a1         = (const float*)d_in[6];
    const float* b_a1         = (const float*)d_in[7];
    const float* W_a2         = (const float*)d_in[8];
    const float* b_a2         = (const float*)d_in[9];
    const float* W_f1         = (const float*)d_in[10];
    const float* b_f1         = (const float*)d_in[11];
    const float* W_f2         = (const float*)d_in[12];
    const float* b_f2         = (const float*)d_in[13];
    const float* ln_g         = (const float*)d_in[14];
    const float* ln_b         = (const float*)d_in[15];

    char* ws = (char*)d_ws;
    unsigned short* pkWlin = (unsigned short*)(ws + 0);          //  45056 B
    unsigned short* pkWa1b = (unsigned short*)(ws + 45056);      //  32768 B
    unsigned short* pkWf1  = (unsigned short*)(ws + 77824);      // 131072 B
    unsigned short* pkWf2  = (unsigned short*)(ws + 208896);     // 131072 B
    float*          sOut   = (float*)(ws + 339968);              // 4 MB
    float*          vOut   = (float*)(ws + 4534272);             // 1.5 MB
    float*          xPre   = (float*)(ws + 6107136);             // 4 MB
    float*          xnF    = (float*)(ws + 10301440);            // 4 MB
    unsigned short* xnB    = (unsigned short*)(ws + 14495744);   // 2 MB
    unsigned short* gBuf   = (unsigned short*)(ws + 16592896);   // 8 MB  (end ~24 MB)

    float* outX = (float*)d_out;                 // (B,N,128)
    float* outV = outX + (size_t)NODES * 128;    // (B,N,16,3)

    pack_weights<<<332, 32, 0, stream>>>(W_lin, W_a1, W_f1, W_f2,
                                         pkWlin, pkWa1b, pkWf1, pkWf2);
    k_lin <<<704,  256, 0, stream>>>(scalar_feats, pkWlin, b_lin, sOut, vOut);
    k_att <<<NODES,256, 0, stream>>>(coords, E_idx, mask, W_a1, W_a2, b_a1, b_a2,
                                     pkWa1b, sOut, vOut, xPre, outV);
    k_ln  <<<NODES,128, 0, stream>>>(xPre, ln_g, ln_b, xnF, xnB);
    k_ffn1<<<2048, 256, 0, stream>>>(xnB, pkWf1, b_f1, gBuf);
    k_ffn2<<<512,  256, 0, stream>>>(gBuf, pkWf2, b_f2, xnF, outX);
}
